// Graph_Model_56169582297518
// MI455X (gfx1250) — matrix operand order repaired and checked by reading, not compiled, not run on hardware
//
#include <hip/hip_runtime.h>
#include <math.h>

// ---------------------------------------------------------------------------
// Model constants (from the reference)
// ---------------------------------------------------------------------------
constexpr int D1   = 1992;           // feature dim
constexpr int EFD  = 12;             // edge feats
constexpr int NBAT = 4;              // graphs per batch
constexpr int PN   = 128;            // protein nodes / graph
constexpr int RN   = 96;             // rna nodes / graph
constexpr int KNN  = 12;             // in-edges per node
constexpr int NIG  = PN + RN;        // 224 inter nodes / graph
constexpr int NI   = NBAT * NIG;     // 896
constexpr int NP   = NBAT * PN;      // 512
constexpr int NR   = NBAT * RN;      // 384
constexpr int EI   = NI * KNN;       // 10752
constexpr int EP   = NP * KNN;       // 6144
constexpr int ER   = NR * KNN;       // 4608
constexpr int FIN  = 2 * D1 + 1 + EFD; // 3997  edge-MLP input width
constexpr int FINP = 4000;           // padded pitch (16B-aligned rows)
constexpr int D2   = 2 * D1;         // 3984
constexpr int D3   = 3 * D1;         // 5976
constexpr int D4   = 4 * D1;         // 7968

typedef __attribute__((ext_vector_type(16))) __bf16 v16bf;
typedef __attribute__((ext_vector_type(8)))  float  v8f;

__device__ __forceinline__ unsigned pack2bf(float x, float y) {
  union { float f; unsigned u; } a, b; a.f = x; b.f = y;
  unsigned rx = a.u + 0x7fffu + ((a.u >> 16) & 1u);   // RNE
  unsigned ry = b.u + 0x7fffu + ((b.u >> 16) & 1u);
  return (rx >> 16) | (ry & 0xffff0000u);
}
__device__ __forceinline__ float sigmf(float x) { return 1.f / (1.f + __expf(-x)); }

// ---------------------------------------------------------------------------
// GEMM: Out[M,N] = act(A[M,K] * B + bias)   (bf16 WMMA core, f32 accumulate)
//   bMode: 0 = B f32 [K,N] row-major
//          1 = B f32 [N,K] row-major (A*B^T)
//          2 = B packed-bf16 dwords [ceil(K/2), N] (dword = {bf16(k), bf16(k+1)})
//   act: 0=none 1=silu 2=relu 3=sigmoid 4=tanh
// Tile 64x128x32, 8 waves (2x4), wave = 32x32 via 2x2 bf16 WMMA frags.
// Double-buffered packed-dword LDS, one barrier per K-step.
// A side: register-staged b128 loads + f32->bf16 pack.
// B side (mode 2, interior tiles): GLOBAL_LOAD_ASYNC_TO_LDS_B32 direct to the
//   transposed LDS layout (per-lane dsaddr), drained with s_wait_asynccnt —
//   no VGPR staging, no conversion. Other modes / edge tiles: register path.
// ---------------------------------------------------------------------------
#define BM 64
#define BN 128
#define BK 32
#define LPW 17   // dwords per LDS row: 16 data + 1 pad (bank-conflict stagger)

__global__ __launch_bounds__(256)
void gemm_wmma_kernel(const float* __restrict__ A, const void* __restrict__ Bv,
                      const float* __restrict__ bias, float* __restrict__ Out,
                      int M, int N, int Kd, int lda, int ldb, int ldo,
                      int act, int bMode) {
  __shared__ unsigned sA[2][BM * LPW];
  __shared__ unsigned sB[2][BN * LPW];

  const int tid  = threadIdx.x;
  const int tM   = blockIdx.y * BM;
  const int tN   = blockIdx.x * BN;
  const int w    = tid >> 5, lane = tid & 31;
  const int wm   = w >> 2,   wn   = w & 3;      // 2x4 wave grid
  const int half = lane >> 4, rr  = lane & 15;

  const bool aFastB = (tM + BM <= M) && ((lda & 3) == 0);
  const bool bFastB = (tN + BN <= N) && (bMode == 2 || ((ldb & 3) == 0));

  float aR[8];                                    // staged A: 2 quads / thread
  union BS { float f[16]; unsigned u[8]; } bS;    // staged B chunk (reg path)

  auto load_A = [&](int kt) {
    const int k0 = kt * BK;
    if (aFastB && (k0 + BK <= Kd)) {
      #pragma unroll
      for (int i = 0; i < 2; ++i) {
        int q = i * 256 + tid, r = q >> 3, qk = (q & 7) << 2;
        float4 v = *(const float4*)(A + (long)(tM + r) * lda + k0 + qk);
        aR[i*4+0] = v.x; aR[i*4+1] = v.y; aR[i*4+2] = v.z; aR[i*4+3] = v.w;
      }
    } else {
      #pragma unroll
      for (int i = 0; i < 2; ++i) {
        int q = i * 256 + tid, r = q >> 3, qk = (q & 7) << 2, gm = tM + r;
        #pragma unroll
        for (int j = 0; j < 4; ++j) {
          int gk = k0 + qk + j;
          aR[i*4+j] = (gm < M && gk < Kd) ? A[(long)gm * lda + gk] : 0.f;
        }
      }
    }
  };
  auto store_A = [&](int buf) {
    #pragma unroll
    for (int i = 0; i < 2; ++i) {
      int q = i * 256 + tid, r = q >> 3, qk = (q & 7) << 2;
      unsigned* dst = &sA[buf][r * LPW + (qk >> 1)];
      dst[0] = pack2bf(aR[i*4+0], aR[i*4+1]);
      dst[1] = pack2bf(aR[i*4+2], aR[i*4+3]);
    }
  };

  // returns true if B chunk is held in registers (needs store_B), false if it
  // was issued as async copies straight into LDS buffer (kt&1).
  auto load_B = [&](int kt) -> bool {
    const int k0 = kt * BK;
    const bool kFull = (k0 + BK <= Kd);
    if (bMode == 2) {
      const unsigned* bP = (const unsigned*)Bv;
      const int kp0 = kt * (BK / 2);
      if (bFastB && kFull) {
        // async global->LDS: 2048 dwords, 8 per lane, per-lane dsaddr gives
        // the [n][k]-transposed layout directly; tracked by ASYNCcnt.
        const int buf = kt & 1;
        #pragma unroll
        for (int i = 0; i < 8; ++i) {
          int t2 = i * 256 + tid;
          int kpl = t2 >> 7, n = t2 & 127;          // lanes sweep n: coalesced
          unsigned ldsOff = (unsigned)(size_t)(&sB[buf][n * LPW + kpl]);
          const unsigned* g = bP + (long)(kp0 + kpl) * ldb + tN + n;
          asm volatile("global_load_async_to_lds_b32 %0, %1, off"
                       :: "v"(ldsOff), "v"(g) : "memory");
        }
        return false;
      }
      const int KP = (Kd + 1) >> 1;
      #pragma unroll
      for (int i = 0; i < 2; ++i) {
        int t2 = i * 256 + tid, kpl = t2 >> 5, nq = (t2 & 31) << 2;
        #pragma unroll
        for (int j = 0; j < 4; ++j) {
          int gn = tN + nq + j;
          bS.u[i*4+j] = (kp0 + kpl < KP && gn < N)
                      ? bP[(long)(kp0 + kpl) * ldb + gn] : 0u;
        }
      }
      return true;
    } else if (bMode == 0) {                // f32 [K,N]
      const float* Bm = (const float*)Bv;
      if (bFastB && kFull) {
        #pragma unroll
        for (int i = 0; i < 2; ++i) {       // rows k,k+1 x 4 cols
          int t2 = i * 256 + tid, kp = t2 >> 5, nq = (t2 & 31) << 2, k = kp * 2;
          float4 v0 = *(const float4*)(Bm + (long)(k0 + k)     * ldb + tN + nq);
          float4 v1 = *(const float4*)(Bm + (long)(k0 + k + 1) * ldb + tN + nq);
          bS.f[i*8+0]=v0.x; bS.f[i*8+1]=v0.y; bS.f[i*8+2]=v0.z; bS.f[i*8+3]=v0.w;
          bS.f[i*8+4]=v1.x; bS.f[i*8+5]=v1.y; bS.f[i*8+6]=v1.z; bS.f[i*8+7]=v1.w;
        }
      } else {
        #pragma unroll
        for (int i = 0; i < 2; ++i) {
          int t2 = i * 256 + tid, kp = t2 >> 5, nq = (t2 & 31) << 2, k = k0 + kp * 2;
          #pragma unroll
          for (int j = 0; j < 4; ++j) {
            int gn = tN + nq + j;
            bS.f[i*8+j]   = (k     < Kd && gn < N) ? Bm[(long)k       * ldb + gn] : 0.f;
            bS.f[i*8+4+j] = (k + 1 < Kd && gn < N) ? Bm[(long)(k + 1) * ldb + gn] : 0.f;
          }
        }
      }
      return true;
    } else {                                // bMode 1: f32 [N,K]  (A*B^T)
      const float* Bm = (const float*)Bv;
      if (bFastB && kFull) {
        #pragma unroll
        for (int i = 0; i < 4; ++i) {       // 4 k per n
          int t2 = i * 256 + tid, n = t2 >> 3, kq = (t2 & 7) << 2;
          float4 v = *(const float4*)(Bm + (long)(tN + n) * ldb + k0 + kq);
          bS.f[i*4+0]=v.x; bS.f[i*4+1]=v.y; bS.f[i*4+2]=v.z; bS.f[i*4+3]=v.w;
        }
      } else {
        #pragma unroll
        for (int i = 0; i < 4; ++i) {
          int t2 = i * 256 + tid, n = tN + (t2 >> 3), kq = k0 + ((t2 & 7) << 2);
          #pragma unroll
          for (int j = 0; j < 4; ++j)
            bS.f[i*4+j] = (n < N && kq + j < Kd) ? Bm[(long)n * ldb + kq + j] : 0.f;
        }
      }
      return true;
    }
  };

  auto store_B = [&](int buf) {
    if (bMode == 2) {
      #pragma unroll
      for (int i = 0; i < 2; ++i) {
        int t2 = i * 256 + tid, kpl = t2 >> 5, nq = (t2 & 31) << 2;
        #pragma unroll
        for (int j = 0; j < 4; ++j)          // already packed: direct copy
          sB[buf][(nq + j) * LPW + kpl] = bS.u[i*4+j];
      }
    } else if (bMode == 0) {
      #pragma unroll
      for (int i = 0; i < 2; ++i) {
        int t2 = i * 256 + tid, kp = t2 >> 5, nq = (t2 & 31) << 2;
        #pragma unroll
        for (int j = 0; j < 4; ++j)          // pack (k,k+1) pairs
          sB[buf][(nq + j) * LPW + kp] = pack2bf(bS.f[i*8+j], bS.f[i*8+4+j]);
      }
    } else {
      #pragma unroll
      for (int i = 0; i < 4; ++i) {
        int t2 = i * 256 + tid, n = t2 >> 3, kq = (t2 & 7) << 2;
        sB[buf][n * LPW + (kq >> 1)]     = pack2bf(bS.f[i*4+0], bS.f[i*4+1]);
        sB[buf][n * LPW + (kq >> 1) + 1] = pack2bf(bS.f[i*4+2], bS.f[i*4+3]);
      }
    }
  };

  v8f acc[2][2];
  #pragma unroll
  for (int i = 0; i < 2; ++i)
    #pragma unroll
    for (int j = 0; j < 2; ++j)
      #pragma unroll
      for (int e = 0; e < 8; ++e) acc[i][j][e] = 0.f;

  const int nk = (Kd + BK - 1) / BK;

  // prologue: fill buffer 0
  load_A(0); store_A(0);
  if (load_B(0)) store_B(0);
  else asm volatile("s_wait_asynccnt 0x0" ::: "memory");

  union Frag { v16bf v; unsigned u[8]; };

  for (int kt = 0; kt < nk; ++kt) {
    __syncthreads();                       // LDS buf (kt&1) ready for all waves
    const int cur = kt & 1;
    const bool more = (kt + 1 < nk);
    bool bInRegs = false;
    if (more) {
      load_A(kt + 1);                      // global/async ops overlap WMMA below
      bInRegs = load_B(kt + 1);
    }

    // ---- fragments in contiguous k order, same for A and B (packed dwords) ----
    // A 16x32 bf16: dword p -> k/2 = half*8 + p,             row = lane%16
    // B 32x16 bf16: dword p -> k/2 = half*8 + p,             col = lane%16
    Frag af[2], bf_[2];
    #pragma unroll
    for (int mi = 0; mi < 2; ++mi) {
      int base = (wm * 32 + mi * 16 + rr) * LPW;
      #pragma unroll
      for (int p = 0; p < 8; ++p)
        af[mi].u[p] = sA[cur][base + (half << 3) + p];
    }
    #pragma unroll
    for (int ni = 0; ni < 2; ++ni) {
      int base = (wn * 32 + ni * 16 + rr) * LPW + (half << 3);
      #pragma unroll
      for (int p = 0; p < 8; ++p)
        bf_[ni].u[p] = sB[cur][base + p];
    }
    #pragma unroll
    for (int mi = 0; mi < 2; ++mi)
      #pragma unroll
      for (int ni = 0; ni < 2; ++ni)
        acc[mi][ni] = __builtin_amdgcn_wmma_f32_16x16x32_bf16(
            false, af[mi].v, false, bf_[ni].v, (short)0, acc[mi][ni], false, false);

    if (more) {
      store_A((kt + 1) & 1);               // other buffer: prior readers done
      if (bInRegs) store_B((kt + 1) & 1);
      else asm volatile("s_wait_asynccnt 0x0" ::: "memory");  // drain own async
    }
  }

  // ---- epilogue: C layout VGPR v -> row = v + 8*half, col = lane%16 ----
  #pragma unroll
  for (int mi = 0; mi < 2; ++mi)
    #pragma unroll
    for (int ni = 0; ni < 2; ++ni) {
      int gc = tN + wn * 32 + ni * 16 + rr;
      float bv = (bias && gc < N) ? bias[gc] : 0.f;
      #pragma unroll
      for (int v = 0; v < 8; ++v) {
        int gr = tM + wm * 32 + mi * 16 + half * 8 + v;
        if (gr < M && gc < N) {
          float x = acc[mi][ni][v] + bv;
          if      (act == 1) x = x * sigmf(x);
          else if (act == 2) x = fmaxf(x, 0.f);
          else if (act == 3) x = sigmf(x);
          else if (act == 4) x = tanhf(x);
          Out[(long)gr * ldo + gc] = x;
        }
      }
    }
}

// ---------------------------------------------------------------------------
// Weight pre-conversion: f32 [K,N] -> packed bf16 dwords [ceil(K/2), N]
// ---------------------------------------------------------------------------
__global__ void cvt_w_kernel(const float* __restrict__ W, unsigned* __restrict__ P,
                             int K, int N) {
  long idx = (long)blockIdx.x * 256 + threadIdx.x;
  long total = (long)((K + 1) >> 1) * N;
  if (idx >= total) return;
  int  n  = (int)(idx % N);
  long kp = idx / N;
  float lo = W[kp * 2 * (long)N + n];
  float hi = (kp * 2 + 1 < K) ? W[(kp * 2 + 1) * (long)N + n] : 0.f;
  P[kp * (long)N + n] = pack2bf(lo, hi);
}

// ---------------------------------------------------------------------------
// Elementwise / reduction scaffolding kernels
// ---------------------------------------------------------------------------

// f[e] = [h[src[e]], h[dst[e]], radial(e), ef[e]]  (FIN cols, FINP pitch)
__global__ void build_edge_f_kernel(const float* __restrict__ h,
                                    const float* __restrict__ coord,
                                    const float* __restrict__ ef,
                                    const int* __restrict__ src,
                                    const int* __restrict__ dst,
                                    float* __restrict__ f) {
  int e = blockIdx.x;
  int s = src[e], d = dst[e];
  float dx = coord[s * 3 + 0] - coord[d * 3 + 0];
  float dy = coord[s * 3 + 1] - coord[d * 3 + 1];
  float dz = coord[s * 3 + 2] - coord[d * 3 + 2];
  float radial = dx * dx + dy * dy + dz * dz;
  float* fr = f + (long)e * FINP;
  for (int c = threadIdx.x; c < FIN; c += blockDim.x) {
    float v;
    if      (c < D1)      v = h[(long)s * D1 + c];
    else if (c < 2 * D1)  v = h[(long)d * D1 + (c - D1)];
    else if (c == 2 * D1) v = radial;
    else                  v = ef[(long)e * EFD + (c - 2 * D1 - 1)];
    fr[c] = v;
  }
}

// nc[n] = [h[n], sum_{j<12} m[n*12+j]]   (dst[e] == e/12 by construction)
__global__ void agg_concat_kernel(const float* __restrict__ h,
                                  const float* __restrict__ m,
                                  float* __restrict__ nc) {
  int d = blockIdx.x * 256 + threadIdx.x;
  int n = blockIdx.y;
  if (d >= D1) return;
  float s = 0.f;
  #pragma unroll
  for (int j = 0; j < KNN; ++j) s += m[((long)n * KNN + j) * D1 + d];
  nc[(long)n * D2 + d]      = h[(long)n * D1 + d];
  nc[(long)n * D2 + D1 + d] = s;
}

// out[r] = dot(A[r,:K], w) + bias[0]
__global__ void rowdot_kernel(const float* __restrict__ A, int lda,
                              const float* __restrict__ w,
                              const float* __restrict__ bias,
                              float* __restrict__ out, int K) {
  __shared__ float red[256];
  int r = blockIdx.x, tid = threadIdx.x;
  float s = 0.f;
  for (int i = tid; i < K; i += 256) s += A[(long)r * lda + i] * w[i];
  red[tid] = s; __syncthreads();
  for (int st = 128; st > 0; st >>= 1) { if (tid < st) red[tid] += red[tid + st]; __syncthreads(); }
  if (tid == 0) out[r] = red[0] + (bias ? bias[0] : 0.f);
}

// per-graph softmax over n contiguous values
__global__ void softmax_seg_kernel(const float* __restrict__ sc, float* __restrict__ gate, int n) {
  __shared__ float red[256];
  int b = blockIdx.x, tid = threadIdx.x;
  const float* s = sc + (long)b * n;
  float m = -1e30f;
  for (int i = tid; i < n; i += 256) m = fmaxf(m, s[i]);
  red[tid] = m; __syncthreads();
  for (int st = 128; st > 0; st >>= 1) { if (tid < st) red[tid] = fmaxf(red[tid], red[tid + st]); __syncthreads(); }
  m = red[0]; __syncthreads();
  float sum = 0.f;
  for (int i = tid; i < n; i += 256) sum += __expf(s[i] - m);
  red[tid] = sum; __syncthreads();
  for (int st = 128; st > 0; st >>= 1) { if (tid < st) red[tid] += red[tid + st]; __syncthreads(); }
  float inv = 1.f / red[0];
  for (int i = tid; i < n; i += 256) gate[(long)b * n + i] = __expf(s[i] - m) * inv;
}

// out[b,d] = sum_i gate[b*n+i] * x[(b*n+i), d]
__global__ void weighted_pool_kernel(const float* __restrict__ gate,
                                     const float* __restrict__ x,
                                     float* __restrict__ out, int n) {
  int d = blockIdx.x * 256 + threadIdx.x, b = blockIdx.y;
  if (d >= D1) return;
  float s = 0.f;
  for (int i = 0; i < n; ++i) s += gate[b * n + i] * x[((long)(b * n + i)) * D1 + d];
  out[(long)b * D1 + d] = s;
}

// dst row r (b-major, T per graph) = src row b*224 + off + t
__global__ void gather_seq_kernel(const float* __restrict__ src, float* __restrict__ dst,
                                  int T, int off) {
  int d = blockIdx.x * 256 + threadIdx.x, r = blockIdx.y;
  if (d >= D1) return;
  int b = r / T, t = r - b * T;
  dst[(long)r * D1 + d] = src[((long)b * NIG + off + t) * D1 + d];
}

// GRU gate math for one timestep
__global__ void gru_gate_kernel(const float* __restrict__ gx, const float* __restrict__ gh,
                                float* __restrict__ h, float* __restrict__ outseq,
                                int t, int T) {
  int d = blockIdx.x * 256 + threadIdx.x, b = blockIdx.y;
  if (d >= D1) return;
  long row = (long)b * T + t;
  float xr = gx[row * D3 + d], xz = gx[row * D3 + D1 + d], xn = gx[row * D3 + 2 * D1 + d];
  float hr = gh[(long)b * D3 + d], hz = gh[(long)b * D3 + D1 + d], hn = gh[(long)b * D3 + 2 * D1 + d];
  float r = sigmf(xr + hr), z = sigmf(xz + hz);
  float nt = tanhf(xn + r * hn);
  float hold = h[(long)b * D1 + d];
  float hnew = (1.f - z) * nt + z * hold;
  h[(long)b * D1 + d] = hnew;
  outseq[row * D1 + d] = hnew;
}

__global__ void zero_kernel(float* __restrict__ p, long n) {
  long i = (long)blockIdx.x * 256 + threadIdx.x;
  if (i < n) p[i] = 0.f;
}

// out[b,d] = mean_t seq[(b*T+t), d]
__global__ void mean_seq_kernel(const float* __restrict__ seq, float* __restrict__ out, int T) {
  int d = blockIdx.x * 256 + threadIdx.x, b = blockIdx.y;
  if (d >= D1) return;
  float s = 0.f;
  for (int t = 0; t < T; ++t) s += seq[((long)b * T + t) * D1 + d];
  out[(long)b * D1 + d] = s / (float)T;
}

__global__ void build_quad_kernel(const float* ph, const float* rh,
                                  const float* phe, const float* rhe,
                                  float* __restrict__ quad) {
  int d = blockIdx.x * 256 + threadIdx.x, b = blockIdx.y;
  if (d >= D1) return;
  quad[((long)b * 4 + 0) * D1 + d] = ph [(long)b * D1 + d];
  quad[((long)b * 4 + 1) * D1 + d] = rh [(long)b * D1 + d];
  quad[((long)b * 4 + 2) * D1 + d] = phe[(long)b * D1 + d];
  quad[((long)b * 4 + 3) * D1 + d] = rhe[(long)b * D1 + d];
}

// alpha[b,i,j] = sum_d tanh(q[b,i,d]+k[b,j,d]) * w[d] + wb
__global__ void cap_alpha_kernel(const float* __restrict__ q, const float* __restrict__ k,
                                 const float* __restrict__ w, const float* __restrict__ wb,
                                 float* __restrict__ alpha) {
  __shared__ float red[256];
  int blk = blockIdx.x, tid = threadIdx.x;
  int b = blk >> 4, i = (blk >> 2) & 3, j = blk & 3;
  const float* qi = q + ((long)b * 4 + i) * D1;
  const float* kj = k + ((long)b * 4 + j) * D1;
  float s = 0.f;
  for (int d = tid; d < D1; d += 256) s += tanhf(qi[d] + kj[d]) * w[d];
  red[tid] = s; __syncthreads();
  for (int st = 128; st > 0; st >>= 1) { if (tid < st) red[tid] += red[tid + st]; __syncthreads(); }
  if (tid == 0) alpha[blk] = red[0] + wb[0];
}

// in-place softmax over groups of gsize consecutive values
__global__ void softmax_groups_kernel(float* __restrict__ x, int ngroups, int gsize) {
  int g = threadIdx.x;
  if (g >= ngroups) return;
  float* p = x + (long)g * gsize;
  float m = -1e30f;
  for (int i = 0; i < gsize; ++i) m = fmaxf(m, p[i]);
  float s = 0.f;
  for (int i = 0; i < gsize; ++i) { p[i] = __expf(p[i] - m); s += p[i]; }
  float inv = 1.f / s;
  for (int i = 0; i < gsize; ++i) p[i] *= inv;
}

// ctx[b,i,d] = sum_j alpha[b,i,j] * quad[b,j,d]
__global__ void cap_context_kernel(const float* __restrict__ alpha,
                                   const float* __restrict__ quad,
                                   float* __restrict__ ctx) {
  int d = blockIdx.x * 256 + threadIdx.x, b = blockIdx.y;
  if (d >= D1) return;
  #pragma unroll
  for (int i = 0; i < 4; ++i) {
    float s = 0.f;
    #pragma unroll
    for (int j = 0; j < 4; ++j)
      s += alpha[b * 16 + i * 4 + j] * quad[((long)b * 4 + j) * D1 + d];
    ctx[((long)b * 4 + i) * D1 + d] = s;
  }
}

// out[b,d] = sum_i sc[b*4+i] * ctx[b,i,d]
__global__ void cap_out_kernel(const float* __restrict__ sc, const float* __restrict__ ctx,
                               float* __restrict__ out) {
  int d = blockIdx.x * 256 + threadIdx.x, b = blockIdx.y;
  if (d >= D1) return;
  float s = 0.f;
  #pragma unroll
  for (int i = 0; i < 4; ++i) s += sc[b * 4 + i] * ctx[((long)b * 4 + i) * D1 + d];
  out[(long)b * D1 + d] = s;
}

// out[r] = [a[r], b2[r]]  (D1 each)
__global__ void concat2_kernel(const float* __restrict__ a, const float* __restrict__ b2,
                               float* __restrict__ out) {
  int d = blockIdx.x * 256 + threadIdx.x, r = blockIdx.y;
  if (d >= D1) return;
  out[(long)r * D2 + d]      = a [(long)r * D1 + d];
  out[(long)r * D2 + D1 + d] = b2[(long)r * D1 + d];
}

// row-wise: S[r,:] /= max(||S[r,:]||_2, 1e-12)
__global__ void row_l2norm_kernel(float* __restrict__ S, int C) {
  __shared__ float red[128];
  int r = blockIdx.x, tid = threadIdx.x;
  float s = 0.f;
  for (int c = tid; c < C; c += 128) { float v = S[(long)r * C + c]; s += v * v; }
  red[tid] = s; __syncthreads();
  for (int st = 64; st > 0; st >>= 1) { if (tid < st) red[tid] += red[tid + st]; __syncthreads(); }
  float inv = 1.f / fmaxf(sqrtf(red[0]), 1e-12f);
  for (int c = tid; c < C; c += 128) S[(long)r * C + c] *= inv;
}

// out[d] = (sum rows a1 + sum rows a2) / (r1+r2), width D2
__global__ void meancat_kernel(const float* __restrict__ a1, int r1,
                               const float* __restrict__ a2, int r2,
                               float* __restrict__ out) {
  int d = blockIdx.x * 256 + threadIdx.x;
  if (d >= D2) return;
  float s = 0.f;
  for (int r = 0; r < r1; ++r) s += a1[(long)r * D2 + d];
  for (int r = 0; r < r2; ++r) s += a2[(long)r * D2 + d];
  out[d] = s / (float)(r1 + r2);
}

// h[b] = [inter_he[b] (D), intra_h_g[b] (D), intra_h_l[b] (2D)]
__global__ void build_h_kernel(const float* __restrict__ he, const float* __restrict__ ihg,
                               const float* __restrict__ ihl, float* __restrict__ out) {
  int c = blockIdx.x * 256 + threadIdx.x, b = blockIdx.y;
  if (c >= D4) return;
  float v;
  if      (c < D1)     v = he [(long)b * D1 + c];
  else if (c < 2 * D1) v = ihg[(long)b * D1 + (c - D1)];
  else                 v = ihl[(long)b * D2 + (c - 2 * D1)];
  out[(long)b * D4 + c] = v;
}

// ---------------------------------------------------------------------------
// Host side
// ---------------------------------------------------------------------------
static inline void gemm(hipStream_t s, const float* A, const void* Bv, const float* bias,
                        float* Out, int M, int N, int Kd, int lda, int ldb, int ldo,
                        int act, int bMode) {
  dim3 g((N + BN - 1) / BN, (M + BM - 1) / BM);
  gemm_wmma_kernel<<<g, 256, 0, s>>>(A, Bv, bias, Out, M, N, Kd, lda, ldb, ldo, act, bMode);
}

extern "C" void kernel_launch(void* const* d_in, const int* in_sizes, int n_in,
                              void* d_out, int out_size, void* d_ws, size_t ws_size,
                              hipStream_t stream) {
  auto F = [&](int i) { return (const float*)d_in[i]; };
  auto I = [&](int i) { return (const int*)d_in[i]; };

  // ---- input slots (setup_inputs order; params pytree alphabetical) ----
  const float* inter_hx    = F(0);
  const float* inter_ex    = F(1);
  const float* inter_coord = F(2);
  const float* pro_hx      = F(3);
  const float* pro_ex      = F(4);
  const float* pro_coord   = F(5);
  const float* rna_hx      = F(6);
  const float* rna_ex      = F(7);
  const float* rna_coord   = F(8);
  // params (9..69)
  const float* att1_b = F(9),  *att1_w = F(10);
  const float* att2_b = F(11), *att2_w = F(12);
  const float* c_b1 = F(13), *c_b2 = F(14), *c_b3 = F(15);
  const float* c_w1 = F(16), *c_w2 = F(17), *c_w3 = F(18);
  const float* capW = F(19), *capWb = F(20), *capWp = F(21), *capWpb = F(22);
  const float* capw = F(23), *capwb = F(24), *capwp = F(25), *capwpb = F(26);
  // egnn dicts, alphabetical: c_b1,c_w1,c_w2,e_b1,e_b2,e_w1,e_w2,n_b1,n_b2,n_w1,n_w2
  const float* egnn0[11], *egnn1[11], *egnn2[11];
  for (int i = 0; i < 11; ++i) egnn0[i] = F(27 + i);
  for (int i = 0; i < 11; ++i) egnn1[i] = F(38 + i);
  for (int i = 0; i < 11; ++i) egnn2[i] = F(49 + i);
  const float* gap_b = F(60), *gap_w = F(61);
  const float* g1_bhh = F(62), *g1_bih = F(63), *g1_whh = F(64), *g1_wih = F(65);
  const float* g2_bhh = F(66), *g2_bih = F(67), *g2_whh = F(68), *g2_wih = F(69);
  const int* inter_src = I(70); const int* inter_dst = I(71);
  const int* pro_src   = I(72); const int* pro_dst   = I(73);
  const int* rna_src   = I(74); const int* rna_dst   = I(75);
  (void)in_sizes; (void)n_in; (void)out_size; (void)ws_size;

  float* out = (float*)d_out;  // [4,7968] then [4,1]

  // ---- workspace allocator ----
  char* wsp = (char*)d_ws;
  auto alloc = [&](size_t nfl) -> float* {
    float* p = (float*)wsp;
    wsp += ((nfl * sizeof(float) + 255) / 256) * 256;
    return p;
  };
  float* fbuf  = alloc((size_t)EI * FINP);  // edge features / reused as m2
  float* mbuf  = alloc((size_t)EI * D1);    // edge m1
  float* ncbuf = alloc((size_t)NI * D2);    // [h, h_agg]
  float* nhbuf = alloc((size_t)NI * D1);    // node hidden
  float* i_hxe = alloc((size_t)NI * D1);
  float* p_hxe = alloc((size_t)NP * D1);
  float* r_hxe = alloc((size_t)NR * D1);
  float* score = alloc(NI);
  float* gate  = alloc(NI);
  float* i_he  = alloc((size_t)NBAT * D1);
  float* p_in  = alloc((size_t)NP * D1);
  float* r_in  = alloc((size_t)NR * D1);
  float* gx_p  = alloc((size_t)NP * D3);
  float* gx_r  = alloc((size_t)NR * D3);
  float* hst   = alloc((size_t)NBAT * D1);
  float* ghb   = alloc((size_t)NBAT * D3);
  float* p_su  = alloc((size_t)NP * D1);
  float* r_su  = alloc((size_t)NR * D1);
  float* p_h   = alloc((size_t)NBAT * D1);
  float* r_h   = alloc((size_t)NBAT * D1);
  float* p_he  = alloc((size_t)NBAT * D1);
  float* r_he  = alloc((size_t)NBAT * D1);
  float* quad  = alloc((size_t)NBAT * 4 * D1);
  float* qcap  = alloc((size_t)NBAT * 4 * D1);
  float* kcap  = alloc((size_t)NBAT * 4 * D1);
  float* alpha = alloc(NBAT * 16);
  float* ctx   = alloc((size_t)NBAT * 4 * D1);
  float* apsc  = alloc(NBAT * 4);
  float* ihg   = alloc((size_t)NBAT * D1);
  float* p_s   = alloc((size_t)NP * D2);
  float* r_s   = alloc((size_t)NR * D2);
  float* qbuf  = alloc((size_t)PN * D2);
  float* kbuf  = alloc((size_t)PN * D2);
  float* sbuf  = alloc((size_t)PN * PN);
  float* abuf1 = alloc((size_t)PN * D2);
  float* abuf2 = alloc((size_t)RN * D2);
  float* ihl   = alloc((size_t)NBAT * D2);
  float* z1    = alloc((size_t)NBAT * D4);
  float* z2    = alloc((size_t)NBAT * D2);

  // ---- pre-convert heavily reused weights to packed bf16 (one cheap pass) ----
  auto cvtw = [&](const float* W, int K, int N) -> const unsigned* {
    unsigned* P = (unsigned*)alloc((size_t)((K + 1) / 2) * N);
    long total = (long)((K + 1) / 2) * N;
    cvt_w_kernel<<<(unsigned)((total + 255) / 256), 256, 0, stream>>>(W, P, K, N);
    return P;
  };
  const unsigned* ew1_16[3]; const unsigned* ew2_16[3];
  const unsigned* nw1_16[3]; const unsigned* nw2_16[3];
  const float* const* eg[3] = { egnn0, egnn1, egnn2 };
  for (int g = 0; g < 3; ++g) {
    ew1_16[g] = cvtw(eg[g][5], FIN, D1);
    ew2_16[g] = cvtw(eg[g][6], D1, D1);
    nw1_16[g] = cvtw(eg[g][9], D2, D1);
    nw2_16[g] = cvtw(eg[g][10], D1, D1);
  }
  const unsigned* w1ih16 = cvtw(g1_wih, D1, D3);
  const unsigned* w1hh16 = cvtw(g1_whh, D1, D3);
  const unsigned* w2ih16 = cvtw(g2_wih, D1, D3);
  const unsigned* w2hh16 = cvtw(g2_whh, D1, D3);
  const unsigned* a1w16  = cvtw(att1_w, D2, D2);
  const unsigned* a2w16  = cvtw(att2_w, D2, D2);
  // single-use weights (cap.W/Wp, c_w1/c_w2) stay f32: conversion would add traffic

  const int gD  = (D1 + 255) / 256;
  const int gD2 = (D2 + 255) / 256;
  const int gD4 = (D4 + 255) / 256;

  // ---- EGNN conv (coord path is dead code in the reference: only h_out used) ----
  auto egnn = [&](const float* h, const float* coord, const float* ef,
                  const int* src, const int* dst, int E, int Nn,
                  const float* const* W, int g, float* hxe) {
    // W: 0=c_b1 1=c_w1 2=c_w2 3=e_b1 4=e_b2 5=e_w1 6=e_w2 7=n_b1 8=n_b2 9=n_w1 10=n_w2
    build_edge_f_kernel<<<E, 256, 0, stream>>>(h, coord, ef, src, dst, fbuf);
    gemm(stream, fbuf, ew1_16[g], W[3], mbuf, E, D1, FIN, FINP, D1, D1, /*silu*/1, 2);
    gemm(stream, mbuf, ew2_16[g], W[4], fbuf, E, D1, D1, D1, D1, D1, /*silu*/1, 2);  // m2
    agg_concat_kernel<<<dim3(gD, Nn), 256, 0, stream>>>(h, fbuf, ncbuf);
    gemm(stream, ncbuf, nw1_16[g], W[7], nhbuf, Nn, D1, D2, D2, D1, D1, /*silu*/1, 2);
    gemm(stream, nhbuf, nw2_16[g], W[8], hxe, Nn, D1, D1, D1, D1, D1, 0, 2);
  };

  // 1) inter-graph EGNN + GlobalAttentionPooling
  egnn(inter_hx, inter_coord, inter_ex, inter_src, inter_dst, EI, NI, egnn0, 0, i_hxe);
  rowdot_kernel<<<NI, 256, 0, stream>>>(i_hxe, D1, gap_w, gap_b, score, D1);
  softmax_seg_kernel<<<NBAT, 256, 0, stream>>>(score, gate, NIG);
  weighted_pool_kernel<<<dim3(gD, NBAT), 256, 0, stream>>>(gate, i_hxe, i_he, NIG);

  // 2) GRUs over original inter node features (pro rows, then rna rows)
  gather_seq_kernel<<<dim3(gD, NP), 256, 0, stream>>>(inter_hx, p_in, PN, 0);
  gather_seq_kernel<<<dim3(gD, NR), 256, 0, stream>>>(inter_hx, r_in, RN, PN);
  auto gru = [&](const float* in, float* gx, float* outseq, int T,
                 const unsigned* wih, const float* bih,
                 const unsigned* whh, const float* bhh) {
    gemm(stream, in, wih, bih, gx, NBAT * T, D3, D1, D1, D3, D3, 0, 2);  // all-step gx
    zero_kernel<<<(NBAT * D1 + 255) / 256, 256, 0, stream>>>(hst, (long)NBAT * D1);
    for (int t = 0; t < T; ++t) {
      gemm(stream, hst, whh, bhh, ghb, NBAT, D3, D1, D1, D3, D3, 0, 2);
      gru_gate_kernel<<<dim3(gD, NBAT), 256, 0, stream>>>(gx, ghb, hst, outseq, t, T);
    }
  };
  gru(p_in, gx_p, p_su, PN, w1ih16, g1_bih, w1hh16, g1_bhh);
  gru(r_in, gx_r, r_su, RN, w2ih16, g2_bih, w2hh16, g2_bhh);
  mean_seq_kernel<<<dim3(gD, NBAT), 256, 0, stream>>>(p_su, p_h, PN);
  mean_seq_kernel<<<dim3(gD, NBAT), 256, 0, stream>>>(r_su, r_h, RN);

  // 3) pro / rna graph EGNNs + means
  egnn(pro_hx, pro_coord, pro_ex, pro_src, pro_dst, EP, NP, egnn1, 1, p_hxe);
  mean_seq_kernel<<<dim3(gD, NBAT), 256, 0, stream>>>(p_hxe, p_he, PN);
  egnn(rna_hx, rna_coord, rna_ex, rna_src, rna_dst, ER, NR, egnn2, 2, r_hxe);
  mean_seq_kernel<<<dim3(gD, NBAT), 256, 0, stream>>>(r_hxe, r_he, RN);

  // 4) graph-level context attention over 4 summary vectors
  build_quad_kernel<<<dim3(gD, NBAT), 256, 0, stream>>>(p_h, r_h, p_he, r_he, quad);
  gemm(stream, quad, capW,  capWb,  qcap, NBAT * 4, D1, D1, D1, D1, D1, 0, 0);
  gemm(stream, quad, capWp, capWpb, kcap, NBAT * 4, D1, D1, D1, D1, D1, 0, 0);
  cap_alpha_kernel<<<NBAT * 16, 256, 0, stream>>>(qcap, kcap, capw, capwb, alpha);
  softmax_groups_kernel<<<1, 64, 0, stream>>>(alpha, NBAT * 4, 4);   // softmax over j
  cap_context_kernel<<<dim3(gD, NBAT), 256, 0, stream>>>(alpha, quad, ctx);
  rowdot_kernel<<<NBAT * 4, 256, 0, stream>>>(ctx, D1, capwp, capwpb, apsc, D1);
  softmax_groups_kernel<<<1, 64, 0, stream>>>(apsc, NBAT, 4);        // softmax over i
  cap_out_kernel<<<dim3(gD, NBAT), 256, 0, stream>>>(apsc, ctx, ihg);

  // 5) node-level cross attention pooling
  concat2_kernel<<<dim3(gD, NP), 256, 0, stream>>>(p_su, p_hxe, p_s);
  concat2_kernel<<<dim3(gD, NR), 256, 0, stream>>>(r_su, r_hxe, r_s);
  for (int b = 0; b < NBAT; ++b) {
    const float* ps = p_s + (size_t)b * PN * D2;
    const float* rs = r_s + (size_t)b * RN * D2;
    // att1: prim=pro, sec=rna
    gemm(stream, ps, a1w16, att1_b, qbuf, PN, D2, D2, D2, D2, D2, 0, 2);
    gemm(stream, rs, a1w16, att1_b, kbuf, RN, D2, D2, D2, D2, D2, 0, 2);
    gemm(stream, qbuf, kbuf, nullptr, sbuf, PN, RN, D2, D2, D2, RN, /*relu*/2, /*B^T*/1);
    row_l2norm_kernel<<<PN, 128, 0, stream>>>(sbuf, RN);
    gemm(stream, sbuf, rs, nullptr, abuf1, PN, D2, RN, RN, D2, D2, 0, 0);
    // att2: prim=rna, sec=pro
    gemm(stream, rs, a2w16, att2_b, qbuf, RN, D2, D2, D2, D2, D2, 0, 2);
    gemm(stream, ps, a2w16, att2_b, kbuf, PN, D2, D2, D2, D2, D2, 0, 2);
    gemm(stream, qbuf, kbuf, nullptr, sbuf, RN, PN, D2, D2, D2, PN, /*relu*/2, /*B^T*/1);
    row_l2norm_kernel<<<RN, 128, 0, stream>>>(sbuf, PN);
    gemm(stream, sbuf, ps, nullptr, abuf2, RN, D2, PN, PN, D2, D2, 0, 0);
    meancat_kernel<<<gD2, 256, 0, stream>>>(abuf1, PN, abuf2, RN, ihl + (size_t)b * D2);
  }

  // 6) assemble h -> d_out, classifier head
  build_h_kernel<<<dim3(gD4, NBAT), 256, 0, stream>>>(i_he, ihg, ihl, out);
  gemm(stream, out, c_w1, c_b1, z1, NBAT, D4, D4, D4, D4, D4, /*relu*/2, 0);
  gemm(stream, z1, c_w2, c_b2, z2, NBAT, D2, D4, D4, D2, D2, 0, 0);
  rowdot_kernel<<<NBAT, 256, 0, stream>>>(z2, D2, c_w3, c_b3, out + (size_t)NBAT * D4, D2);
}
